// GATModel_test_56581899158036
// MI455X (gfx1250) — matrix operand order repaired and checked by reading, not compiled, not run on hardware
//
#include <hip/hip_runtime.h>

// ============================================================================
// CDNA5 (gfx1250) implementation of the GAT reference model.
// Dense GEMMs: v_wmma_f32_16x16x32_bf16 (wave32 WMMA), double-buffered LDS,
// A tiles staged with global_load_async_to_lds_b128 (ASYNCcnt-tracked DMA).
// ============================================================================

typedef __attribute__((ext_vector_type(16))) __bf16 v16bf;
typedef __attribute__((ext_vector_type(8)))  float  v8f;

union FragAB { v16bf v; uint4 u[2]; };

// ---------------------------------------------------------------------------
// WMMA GEMM: C[M,N] = A[M,K] @ B[K,N] + bias (A,B bf16 row-major, C fp32/bf16)
// Block tile 128x128, K-step 32, double buffered. 8 waves (4 M x 2 N),
// each wave computes 32x64 via 2x4 v_wmma accumulators.
// ---------------------------------------------------------------------------
#define TM 128
#define TN 128
#define TK 32
#define LDSS (TK + 8)   // padded stride (40 bf16 = 20 dwords) -> conflict-free

template <bool FULLN>
__global__ __launch_bounds__(256) void gemm_bf16_wmma(
    const __bf16* __restrict__ A, const __bf16* __restrict__ B,
    const float* __restrict__ bias, float* __restrict__ Cf,
    __bf16* __restrict__ Cb, int M, int N, int K, int nOff)
{
  __shared__ __align__(16) __bf16 sA[2][TM][LDSS];
  __shared__ __align__(16) __bf16 sBT[2][TN][LDSS];  // B transposed [n][k]

  const int tid  = threadIdx.x;
  const int lane = tid & 31;
  const int wid  = tid >> 5;
  const int wm   = wid & 3;          // wave tile row (0..3), 32 rows each
  const int wn   = wid >> 2;         // wave tile col (0..1), 64 cols each
  const int bm   = blockIdx.y * TM;
  const int bn   = nOff + blockIdx.x * TN;
  const int half = lane >> 4;        // 0: lanes 0-15, 1: lanes 16-31
  const int l15  = lane & 15;

  const v8f vzero = {0.f,0.f,0.f,0.f,0.f,0.f,0.f,0.f};
  v8f acc[2][4];
#pragma unroll
  for (int i = 0; i < 2; ++i)
#pragma unroll
    for (int j = 0; j < 4; ++j) acc[i][j] = vzero;

  // ---- A tile staging: async DMA directly into LDS (no VGPR round-trip) ----
  auto stageA = [&](int k0, int buf) {
#pragma unroll
    for (int it = 0; it < 2; ++it) {
      int linear = (tid + it * 256) * 8;           // 8 bf16 = 16B per transfer
      int r = linear >> 5, c = linear & 31;
      const __bf16* gp = A + (size_t)(bm + r) * K + (k0 + c);
      unsigned ldsOff = (unsigned)(size_t)&sA[buf][r][c];  // low 32b = LDS offset
      asm volatile("global_load_async_to_lds_b128 %0, %1, off"
                   :: "v"(ldsOff), "v"(gp) : "memory");
    }
  };
  // ---- B tile staging: vector load + transposed store into LDS ----
  auto stageB = [&](int k0, int buf) {
#pragma unroll
    for (int it = 0; it < 2; ++it) {
      int linear = (tid + it * 256) * 8;
      int kr = linear >> 7, nc = linear & 127;
      int gn = bn + nc;
      if (FULLN) {
        uint4 bv = *(const uint4*)(B + (size_t)(k0 + kr) * N + gn);
        const __bf16* be = (const __bf16*)&bv;
#pragma unroll
        for (int j = 0; j < 8; ++j) sBT[buf][nc + j][kr] = be[j];
        if (k0 + TK < K)  // gfx1250 global_prefetch_b8 for the next B tile
          __builtin_prefetch(B + (size_t)(k0 + TK + kr) * N + gn, 0, 1);
      } else {
#pragma unroll
        for (int j = 0; j < 8; ++j) {
          __bf16 e = (gn + j < N) ? B[(size_t)(k0 + kr) * N + gn + j] : (__bf16)0.f;
          sBT[buf][nc + j][kr] = e;
        }
      }
    }
  };

  const int nk = K / TK;
  stageA(0, 0);
  stageB(0, 0);

  for (int ki = 0; ki < nk; ++ki) {
    const int buf = ki & 1;
    asm volatile("s_wait_asynccnt 0" ::: "memory");  // my async stores landed
    __syncthreads();                                 // everyone's tile visible
    if (ki + 1 < nk) {                               // overlap next-tile DMA
      stageA((ki + 1) * TK, buf ^ 1);
      stageB((ki + 1) * TK, buf ^ 1);
    }

    // ---- fragments per the ISA 16-bit layouts:
    // A 16x32: lanes 0-15 -> K {0..7,16..23}, lanes 16-31 -> K {8..15,24..31}
    // B 32x16 (from transposed LDS): lanes 0-15 -> K {0..7,16..23}, lanes 16-31 -> K {8..15,24..31}
    FragAB af[2], bfr[4];
    const int ak = half ? 8 : 0;
#pragma unroll
    for (int i = 0; i < 2; ++i) {
      int row = wm * 32 + i * 16 + l15;
      af[i].u[0] = *(const uint4*)&sA[buf][row][ak];
      af[i].u[1] = *(const uint4*)&sA[buf][row][ak + 16];
    }
    const int bk = half ? 8 : 0;
#pragma unroll
    for (int j = 0; j < 4; ++j) {
      int col = wn * 64 + j * 16 + l15;
      bfr[j].u[0] = *(const uint4*)&sBT[buf][col][bk];
      bfr[j].u[1] = *(const uint4*)&sBT[buf][col][bk + 16];
    }
#pragma unroll
    for (int i = 0; i < 2; ++i)
#pragma unroll
      for (int j = 0; j < 4; ++j)
        acc[i][j] = __builtin_amdgcn_wmma_f32_16x16x32_bf16(
            false, af[i].v, false, bfr[j].v, (short)0, acc[i][j], false, false);
  }

  // ---- epilogue: D layout = VGPR r -> row r (+8 for lanes 16-31), col = lane&15
#pragma unroll
  for (int i = 0; i < 2; ++i) {
    int rbase = bm + wm * 32 + i * 16 + (half ? 8 : 0);
#pragma unroll
    for (int j = 0; j < 4; ++j) {
      int col = bn + wn * 64 + j * 16 + l15;
      if (FULLN || col < N) {
        float bv = bias ? bias[col] : 0.f;
#pragma unroll
        for (int r = 0; r < 8; ++r) {
          float d = acc[i][j][r] + bv;
          size_t idx = (size_t)(rbase + r) * N + col;
          if (Cf) Cf[idx] = d;
          if (Cb) Cb[idx] = (__bf16)d;
        }
      }
    }
  }
}

// ---------------------------------------------------------------------------
// Helpers (wave32)
// ---------------------------------------------------------------------------
__device__ __forceinline__ float waveReduceSum(float v) {
#pragma unroll
  for (int off = 16; off > 0; off >>= 1) v += __shfl_xor(v, off, 32);
  return v;
}
// monotonic float<->int key for atomicMax-based segment max
__device__ __forceinline__ int f2ord(float f) {
  int i = __float_as_int(f);
  return (i >= 0) ? i : (i ^ 0x7fffffff);
}
__device__ __forceinline__ float ord2f(int i) {
  return __int_as_float((i >= 0) ? i : (i ^ 0x7fffffff));
}

// ---------------------------------------------------------------------------
// Per-edge-head logits for TransformerConv: dot(q[dst,h,:], k[src,h,:]) * scale
// one wave per (edge, head) task; C is a multiple of 256 (1024 or 512)
// ---------------------------------------------------------------------------
__global__ __launch_bounds__(256) void tconv_edge_logits(
    const int* __restrict__ ei, const __bf16* __restrict__ q,
    const __bf16* __restrict__ k, float* __restrict__ logits,
    int E, int H, int C, float scale)
{
  int task = blockIdx.x * 8 + (threadIdx.x >> 5);
  if (task >= E * H) return;
  int e = task / H, h = task % H, lane = threadIdx.x & 31;
  int src = ei[e * 2 + 0], dst = ei[e * 2 + 1];
  const __bf16* qp = q + (size_t)dst * H * C + (size_t)h * C;
  const __bf16* kp = k + (size_t)src * H * C + (size_t)h * C;
  float acc = 0.f;
  for (int cb = lane * 8; cb < C; cb += 256) {
    uint4 qu = *(const uint4*)(qp + cb);
    uint4 ku = *(const uint4*)(kp + cb);
    const __bf16* qe = (const __bf16*)&qu;
    const __bf16* ke = (const __bf16*)&ku;
#pragma unroll
    for (int j = 0; j < 8; ++j) acc += (float)qe[j] * (float)ke[j];
  }
  acc = waveReduceSum(acc);
  if (lane == 0) logits[task] = acc * scale;
}

// segment max over dst (edges + optional NS self loops), keys stored as ints
__global__ void seg_max_k(const float* __restrict__ logits, const int* __restrict__ ei,
                          int* __restrict__ m, int E, int NS, int H)
{
  int t = blockIdx.x * blockDim.x + threadIdx.x;
  if (t >= (E + NS) * H) return;
  int e = t / H, h = t % H;
  int d = (e < E) ? ei[e * 2 + 1] : (e - E);
  atomicMax(&m[d * H + h], f2ord(logits[t]));
}

// w = exp(l - m[dst]); logits <- w; s[dst] += w
__global__ void seg_expsum_k(float* __restrict__ logits, const int* __restrict__ ei,
                             const int* __restrict__ m, float* __restrict__ s,
                             int E, int NS, int H)
{
  int t = blockIdx.x * blockDim.x + threadIdx.x;
  if (t >= (E + NS) * H) return;
  int e = t / H, h = t % H;
  int d = (e < E) ? ei[e * 2 + 1] : (e - E);
  float w = __expf(logits[t] - ord2f(m[d * H + h]));
  logits[t] = w;
  atomicAdd(&s[d * H + h], w);
}

// out[dst, hOff + c] += v[src,h,c] * (w/s) * hscale   (wave per (edge,head))
__global__ __launch_bounds__(256) void tconv_scatter(
    const int* __restrict__ ei, const __bf16* __restrict__ v,
    const float* __restrict__ logits, const float* __restrict__ s,
    float* __restrict__ out, int E, int H, int C,
    int outStride, int headStride, float hscale)
{
  int task = blockIdx.x * 8 + (threadIdx.x >> 5);
  if (task >= E * H) return;
  int e = task / H, h = task % H, lane = threadIdx.x & 31;
  int src = ei[e * 2 + 0], dst = ei[e * 2 + 1];
  float alpha = logits[task] / s[dst * H + h] * hscale;
  const __bf16* vp = v + (size_t)src * H * C + (size_t)h * C;
  float* op = out + (size_t)dst * outStride + (size_t)h * headStride;
  for (int cb = lane * 8; cb < C; cb += 256) {
    uint4 vu = *(const uint4*)(vp + cb);
    const __bf16* ve = (const __bf16*)&vu;
#pragma unroll
    for (int j = 0; j < 8; ++j) atomicAdd(&op[cb + j], (float)ve[j] * alpha);
  }
}

// GATv2 logits: leaky_relu(xl[s]+xr[d], 0.2) . att[h]   (self loops included)
__global__ __launch_bounds__(256) void gat_edge_logits(
    const int* __restrict__ ei, const float* __restrict__ xl,
    const float* __restrict__ xr, const float* __restrict__ att,
    float* __restrict__ logits, int E, int NS, int H, int C)
{
  int task = blockIdx.x * 8 + (threadIdx.x >> 5);
  if (task >= (E + NS) * H) return;
  int e = task / H, h = task % H, lane = threadIdx.x & 31;
  int sn = (e < E) ? ei[e * 2 + 0] : (e - E);
  int dn = (e < E) ? ei[e * 2 + 1] : (e - E);
  const float* lp = xl + (size_t)sn * H * C + (size_t)h * C;
  const float* rp = xr + (size_t)dn * H * C + (size_t)h * C;
  const float* ap = att + (size_t)h * C;
  float acc = 0.f;
  for (int c = lane; c < C; c += 32) {
    float v = lp[c] + rp[c];
    v = v > 0.f ? v : 0.2f * v;
    acc += v * ap[c];
  }
  acc = waveReduceSum(acc);
  if (lane == 0) logits[task] = acc;
}

// out[d, c] += xl[s,h,c] * (w/s) / H
__global__ __launch_bounds__(256) void gat_scatter(
    const int* __restrict__ ei, const float* __restrict__ xl,
    const float* __restrict__ logits, const float* __restrict__ s,
    float* __restrict__ out, int E, int NS, int H, int C)
{
  int task = blockIdx.x * 8 + (threadIdx.x >> 5);
  if (task >= (E + NS) * H) return;
  int e = task / H, h = task % H, lane = threadIdx.x & 31;
  int sn = (e < E) ? ei[e * 2 + 0] : (e - E);
  int dn = (e < E) ? ei[e * 2 + 1] : (e - E);
  float alpha = logits[task] / s[dn * H + h] * (1.f / 8.f);
  const float* xp = xl + (size_t)sn * H * C + (size_t)h * C;
  float* op = out + (size_t)dn * C;
  for (int c = lane; c < C; c += 32) atomicAdd(&op[c], xp[c] * alpha);
}

// ---------------------------------------------------------------------------
// Elementwise kernels
// ---------------------------------------------------------------------------
__global__ void cvt_f32_to_bf16(__bf16* __restrict__ d, const float* __restrict__ s, size_t n) {
  size_t i = (size_t)blockIdx.x * 256 + threadIdx.x;
  if (i < n) d[i] = (__bf16)s[i];
}
__global__ void sub_cvt_k(const float* __restrict__ x, const float* __restrict__ h1,
                          __bf16* __restrict__ x2b, size_t n) {
  size_t i = (size_t)blockIdx.x * 256 + threadIdx.x;
  if (i < n) x2b[i] = (__bf16)(x[i] - h1[i]);
}
__global__ void elu_cvt_k(const float* __restrict__ h, __bf16* __restrict__ d, size_t n) {
  size_t i = (size_t)blockIdx.x * 256 + threadIdx.x;
  if (i < n) { float v = h[i]; d[i] = (__bf16)(v > 0.f ? v : (__expf(v) - 1.f)); }
}
__global__ void fill_i32_k(int* p, int v, int n) {
  int i = blockIdx.x * 256 + threadIdx.x; if (i < n) p[i] = v;
}
__global__ void fill_f32_k(float* p, float v, int n) {
  int i = blockIdx.x * 256 + threadIdx.x; if (i < n) p[i] = v;
}
__global__ void bias_rows_k(float* __restrict__ out, const float* __restrict__ b,
                            int rows, int cols) {
  size_t i = (size_t)blockIdx.x * 256 + threadIdx.x;
  if (i < (size_t)rows * cols) out[i] = b[i % cols];
}
__global__ void copy_f32_k(float* __restrict__ d, const float* __restrict__ s, size_t n) {
  size_t i = (size_t)blockIdx.x * 256 + threadIdx.x; if (i < n) d[i] = s[i];
}
__global__ void i32_to_f32_k(float* __restrict__ d, const int* __restrict__ s, size_t n) {
  size_t i = (size_t)blockIdx.x * 256 + threadIdx.x; if (i < n) d[i] = (float)s[i];
}

// ---------------------------------------------------------------------------
// Host launcher
// ---------------------------------------------------------------------------
extern "C" void kernel_launch(void* const* d_in, const int* in_sizes, int n_in,
                              void* d_out, int out_size, void* d_ws, size_t ws_size,
                              hipStream_t stream)
{
  (void)in_sizes; (void)n_in; (void)out_size; (void)ws_size;
  const int Nn = 8192, Ee = 32768, Dd = 1024, NC = 460;
  const int HC1 = 4096, HC2 = 2048, GC = 3680;

  const float* x    = (const float*)d_in[0];
  const float* exps = (const float*)d_in[1];
  const float* expc = (const float*)d_in[2];
  const int*   ei   = (const int*)d_in[3];
  const float* Wq1 = (const float*)d_in[4];  const float* bq1 = (const float*)d_in[5];
  const float* Wk1 = (const float*)d_in[6];  const float* bk1 = (const float*)d_in[7];
  const float* Wv1 = (const float*)d_in[8];  const float* bv1 = (const float*)d_in[9];
  const float* Ws1 = (const float*)d_in[10]; const float* bs1 = (const float*)d_in[11];
  const float* Wq2 = (const float*)d_in[12]; const float* bq2 = (const float*)d_in[13];
  const float* Wk2 = (const float*)d_in[14]; const float* bk2 = (const float*)d_in[15];
  const float* Wv2 = (const float*)d_in[16]; const float* bv2 = (const float*)d_in[17];
  const float* Ws2 = (const float*)d_in[18]; const float* bs2 = (const float*)d_in[19];
  const float* Wl0 = (const float*)d_in[20]; const float* bl0 = (const float*)d_in[21];
  const float* Wr0 = (const float*)d_in[22]; const float* br0 = (const float*)d_in[23];
  const float* att0= (const float*)d_in[24]; const float* b0  = (const float*)d_in[25];

  // ---- workspace linear allocator (~450 MB total) ----
  char* wsp = (char*)d_ws;
  auto alloc = [&](size_t bytes) -> char* {
    char* r = wsp; wsp += (bytes + 255) & ~(size_t)255; return r;
  };
  __bf16* xb   = (__bf16*)alloc((size_t)Nn * Dd * 2);
  __bf16* Wq1b = (__bf16*)alloc((size_t)Dd * HC1 * 2);
  __bf16* Wk1b = (__bf16*)alloc((size_t)Dd * HC1 * 2);
  __bf16* Wv1b = (__bf16*)alloc((size_t)Dd * HC1 * 2);
  __bf16* Ws1b = (__bf16*)alloc((size_t)Dd * Dd  * 2);
  __bf16* Wq2b = (__bf16*)alloc((size_t)Dd * HC2 * 2);
  __bf16* Wk2b = (__bf16*)alloc((size_t)Dd * HC2 * 2);
  __bf16* Wv2b = (__bf16*)alloc((size_t)Dd * HC2 * 2);
  __bf16* Ws2b = (__bf16*)alloc((size_t)Dd * HC2 * 2);
  __bf16* Wl0b = (__bf16*)alloc((size_t)HC2 * GC * 2);
  __bf16* Wr0b = (__bf16*)alloc((size_t)HC2 * GC * 2);
  // big region reused: q1/k1/v1 -> q2/k2/v2 -> xl/xr
  char* big = alloc((size_t)2 * Nn * GC * 4);               // 241 MB
  __bf16* q1 = (__bf16*)big;
  __bf16* k1 = (__bf16*)(big + (size_t)Nn * HC1 * 2);
  __bf16* v1 = (__bf16*)(big + (size_t)2 * Nn * HC1 * 2);
  __bf16* q2 = q1; __bf16* k2 = k1; __bf16* v2 = v1;
  float* xl = (float*)big;
  float* xr = (float*)(big + (size_t)Nn * GC * 4);
  __bf16* x2b = (__bf16*)alloc((size_t)Nn * Dd * 2);
  float*  h   = (float*)alloc((size_t)Nn * HC2 * 4);
  __bf16* heb = (__bf16*)alloc((size_t)Nn * HC2 * 2);
  float* logits  = (float*)alloc((size_t)Ee * 4 * 4);
  int*   mbuf    = (int*)  alloc((size_t)Nn * 8 * 4);
  float* sbuf    = (float*)alloc((size_t)Nn * 8 * 4);
  float* logits0 = (float*)alloc((size_t)(Ee + Nn) * 8 * 4);

  // ---- d_out layout: [h0][exps][exps_c][edge_index][h1] ----
  float* out   = (float*)d_out;
  float* h0o   = out;
  float* expso = out + (size_t)Nn * NC;
  float* expco = expso + (size_t)Nn * NC;
  float* eio   = expco + (size_t)128 * NC;
  float* h1    = eio + (size_t)Ee * 2;     // h1 computed in-place in d_out

  auto cvt = [&](const float* s, __bf16* d, size_t n) {
    cvt_f32_to_bf16<<<(unsigned)((n + 255) / 256), 256, 0, stream>>>(d, s, n);
  };
  auto gemm = [&](const __bf16* A, const __bf16* B, const float* bias,
                  float* Cf, __bf16* Cb, int M, int N, int K) {
    int nfull = N / TN, nrem = N % TN;
    if (nfull)
      gemm_bf16_wmma<true><<<dim3(nfull, M / TM), dim3(256), 0, stream>>>(
          A, B, bias, Cf, Cb, M, N, K, 0);
    if (nrem)
      gemm_bf16_wmma<false><<<dim3(1, M / TM), dim3(256), 0, stream>>>(
          A, B, bias, Cf, Cb, M, N, K, nfull * TN);
  };

  // ---- stage inputs/weights to bf16 ----
  cvt(x, xb, (size_t)Nn * Dd);
  cvt(Wq1, Wq1b, (size_t)Dd * HC1); cvt(Wk1, Wk1b, (size_t)Dd * HC1);
  cvt(Wv1, Wv1b, (size_t)Dd * HC1); cvt(Ws1, Ws1b, (size_t)Dd * Dd);
  cvt(Wq2, Wq2b, (size_t)Dd * HC2); cvt(Wk2, Wk2b, (size_t)Dd * HC2);
  cvt(Wv2, Wv2b, (size_t)Dd * HC2); cvt(Ws2, Ws2b, (size_t)Dd * HC2);
  cvt(Wl0, Wl0b, (size_t)HC2 * GC); cvt(Wr0, Wr0b, (size_t)HC2 * GC);

  // ================= Layer 1: TransformerConv(H=4, C=1024, mean) ============
  gemm(xb, Wq1b, bq1, nullptr, q1, Nn, HC1, Dd);
  gemm(xb, Wk1b, bk1, nullptr, k1, Nn, HC1, Dd);
  gemm(xb, Wv1b, bv1, nullptr, v1, Nn, HC1, Dd);
  gemm(xb, Ws1b, bs1, h1, nullptr, Nn, Dd, Dd);          // h1 = skip (then +=attn)

  int T1 = Ee * 4;
  tconv_edge_logits<<<(T1 + 7) / 8, 256, 0, stream>>>(ei, q1, k1, logits, Ee, 4, 1024, 1.f / 32.f);
  fill_i32_k<<<(Nn * 4 + 255) / 256, 256, 0, stream>>>(mbuf, (int)0x80000000, Nn * 4);
  fill_f32_k<<<(Nn * 4 + 255) / 256, 256, 0, stream>>>(sbuf, 0.f, Nn * 4);
  seg_max_k   <<<(T1 + 255) / 256, 256, 0, stream>>>(logits, ei, mbuf, Ee, 0, 4);
  seg_expsum_k<<<(T1 + 255) / 256, 256, 0, stream>>>(logits, ei, mbuf, sbuf, Ee, 0, 4);
  tconv_scatter<<<(T1 + 7) / 8, 256, 0, stream>>>(ei, v1, logits, sbuf, h1,
                                                  Ee, 4, 1024, 1024, 0, 0.25f);

  // x2 = x - h1 (bf16 for next GEMMs)
  sub_cvt_k<<<((size_t)Nn * Dd + 255) / 256, 256, 0, stream>>>(x, h1, x2b, (size_t)Nn * Dd);

  // ================= Layer 2: TransformerConv(H=4, C=512, concat) ===========
  gemm(x2b, Wq2b, bq2, nullptr, q2, Nn, HC2, Dd);
  gemm(x2b, Wk2b, bk2, nullptr, k2, Nn, HC2, Dd);
  gemm(x2b, Wv2b, bv2, nullptr, v2, Nn, HC2, Dd);
  gemm(x2b, Ws2b, bs2, h, nullptr, Nn, HC2, Dd);         // h = skip (then +=attn)

  tconv_edge_logits<<<(T1 + 7) / 8, 256, 0, stream>>>(ei, q2, k2, logits, Ee, 4, 512, 0.044194174f);
  fill_i32_k<<<(Nn * 4 + 255) / 256, 256, 0, stream>>>(mbuf, (int)0x80000000, Nn * 4);
  fill_f32_k<<<(Nn * 4 + 255) / 256, 256, 0, stream>>>(sbuf, 0.f, Nn * 4);
  seg_max_k   <<<(T1 + 255) / 256, 256, 0, stream>>>(logits, ei, mbuf, Ee, 0, 4);
  seg_expsum_k<<<(T1 + 255) / 256, 256, 0, stream>>>(logits, ei, mbuf, sbuf, Ee, 0, 4);
  tconv_scatter<<<(T1 + 7) / 8, 256, 0, stream>>>(ei, v2, logits, sbuf, h,
                                                  Ee, 4, 512, 2048, 512, 1.0f);

  // heb = bf16(elu(h))
  elu_cvt_k<<<((size_t)Nn * HC2 + 255) / 256, 256, 0, stream>>>(h, heb, (size_t)Nn * HC2);

  // ================= GATv2 (H=8, C=460, mean) ===============================
  gemm(heb, Wl0b, bl0, xl, nullptr, Nn, GC, HC2);
  gemm(heb, Wr0b, br0, xr, nullptr, Nn, GC, HC2);

  bias_rows_k<<<((size_t)Nn * NC + 255) / 256, 256, 0, stream>>>(h0o, b0, Nn, NC);
  int T0 = (Ee + Nn) * 8;
  gat_edge_logits<<<(T0 + 7) / 8, 256, 0, stream>>>(ei, xl, xr, att0, logits0, Ee, Nn, 8, NC);
  fill_i32_k<<<(Nn * 8 + 255) / 256, 256, 0, stream>>>(mbuf, (int)0x80000000, Nn * 8);
  fill_f32_k<<<(Nn * 8 + 255) / 256, 256, 0, stream>>>(sbuf, 0.f, Nn * 8);
  seg_max_k   <<<(T0 + 255) / 256, 256, 0, stream>>>(logits0, ei, mbuf, Ee, Nn, 8);
  seg_expsum_k<<<(T0 + 255) / 256, 256, 0, stream>>>(logits0, ei, mbuf, sbuf, Ee, Nn, 8);
  gat_scatter<<<(T0 + 7) / 8, 256, 0, stream>>>(ei, xl, logits0, sbuf, h0o, Ee, Nn, 8, NC);

  // ================= passthrough outputs ====================================
  copy_f32_k<<<((size_t)Nn * NC + 255) / 256, 256, 0, stream>>>(expso, exps, (size_t)Nn * NC);
  copy_f32_k<<<((size_t)128 * NC + 255) / 256, 256, 0, stream>>>(expco, expc, (size_t)128 * NC);
  i32_to_f32_k<<<((size_t)Ee * 2 + 255) / 256, 256, 0, stream>>>(eio, ei, (size_t)Ee * 2);
}
